// ST_HGAT_68882685493339
// MI455X (gfx1250) — matrix operand order repaired and checked by reading, not compiled, not run on hardware
//
#include <hip/hip_runtime.h>
#include <math.h>

// ST_HGAT on gfx1250: bf16 WMMA for all dense GEMMs (16x32 tile per wave, dual
// accumulators for independent WMMA chains), atomic 3-pass edge softmax.

typedef __attribute__((ext_vector_type(16))) __bf16 v16bf;
typedef __attribute__((ext_vector_type(8)))  float  v8f;

#define HIDW 128
#define HEADSW 4

static __device__ __forceinline__ float lrelu_f(float x, float s){ return x >= 0.f ? x : s * x; }

static __device__ __forceinline__ unsigned short f2bf(float f){
  unsigned u = __float_as_uint(f);
  u += 0x7FFFu + ((u >> 16) & 1u);           // round-to-nearest-even
  return (unsigned short)(u >> 16);
}

// ordered-uint encoding so unsigned atomicMax == float max
static __device__ __forceinline__ unsigned ord_enc(float f){
  unsigned u = __float_as_uint(f);
  return (u & 0x80000000u) ? ~u : (u | 0x80000000u);
}
static __device__ __forceinline__ float ord_dec(unsigned e){
  if (e == 0u) return 0.0f;                  // sentinel: no edges -> m = 0 (matches isfinite guard)
  unsigned u = (e & 0x80000000u) ? (e & 0x7FFFFFFFu) : ~e;
  float f = __uint_as_float(u);
  if (!(fabsf(f) <= 3.402823466e38f)) return 0.0f;
  return f;
}

static __device__ __forceinline__ float softplus_f(float x){
  if (x > 20.f) return x;
  return log1pf(expf(x));
}

// ---------------- encoder: z = lrelu(x@W+b); layernorm; write f32 + bf16 ----------------
__global__ void enc_k(const float* __restrict__ x, int in_dim,
                      const float* __restrict__ W, const float* __restrict__ b,
                      const float* __restrict__ be, const float* __restrict__ g,
                      float* __restrict__ h, unsigned short* __restrict__ hb, int n){
  int node = blockIdx.x;
  int t = threadIdx.x;
  if (node >= n) return;
  __shared__ float xs[16];
  __shared__ float red[HIDW];
  __shared__ float mu_s, var_s;
  if (t < in_dim) xs[t] = x[(size_t)node * in_dim + t];
  __syncthreads();
  float z = b[t];
  for (int k = 0; k < in_dim; k++) z += xs[k] * W[k * HIDW + t];
  z = lrelu_f(z, 0.01f);
  red[t] = z; __syncthreads();
  for (int s = 64; s > 0; s >>= 1){ if (t < s) red[t] += red[t + s]; __syncthreads(); }
  if (t == 0) mu_s = red[0] * (1.0f / HIDW);
  __syncthreads();
  float d = z - mu_s;
  red[t] = d * d; __syncthreads();
  for (int s = 64; s > 0; s >>= 1){ if (t < s) red[t] += red[t + s]; __syncthreads(); }
  if (t == 0) var_s = red[0] * (1.0f / HIDW);
  __syncthreads();
  float hv = d * rsqrtf(var_s + 1e-5f) * g[t] + be[t];
  h [(size_t)node * HIDW + t] = hv;
  hb[(size_t)node * HIDW + t] = f2bf(hv);
}

// ---------------- transpose+convert W[K][N] f32 -> Wt[N][K] bf16 ----------------
__global__ void prep_wt_k(const float* __restrict__ W, unsigned short* __restrict__ Wt,
                          int K, int Nn){
  int id = blockIdx.x * blockDim.x + threadIdx.x;
  if (id >= K * Nn) return;
  int nn = id / K, k = id % K;
  Wt[(size_t)nn * K + k] = f2bf(W[(size_t)k * Nn + nn]);
}

// ---------------- was[k][h] = sum_o W[k][32h+o]*as[h][o]; wad likewise ----------------
__global__ void prep_attvec_k(const float* __restrict__ W, const float* __restrict__ av_s,
                              const float* __restrict__ av_d,
                              float* __restrict__ was, float* __restrict__ wad){
  int id = blockIdx.x * blockDim.x + threadIdx.x;
  if (id >= HIDW * HEADSW) return;
  int k = id >> 2, hh = id & 3;
  float s0 = 0.f, s1 = 0.f;
  const float* wrow = W + (size_t)k * HIDW + hh * 32;
  const float* a0 = av_s + hh * 32;
  const float* a1 = av_d + hh * 32;
  for (int o = 0; o < 32; o++){ s0 += wrow[o] * a0[o]; s1 += wrow[o] * a1[o]; }
  was[k * HEADSW + hh] = s0;
  wad[k * HEADSW + hh] = s1;
}

// ---------------- bf16 WMMA GEMM: C[n x Nn] = A[n x K] * Bt[Nn x K]^T ----------------
// one wave per 16(M) x 32(N) output tile: A fragment shared by two independent
// accumulators -> two interleaved WMMA chains, half the A load traffic.
// K multiple of 32; Nn multiple of 32.
__global__ void gemm_wmma_k(const unsigned short* __restrict__ A,
                            const unsigned short* __restrict__ Bt,
                            float* __restrict__ C, int n, int K, int Nn){
  const __bf16* Ab = reinterpret_cast<const __bf16*>(A);
  const __bf16* Bb = reinterpret_cast<const __bf16*>(Bt);
  int lane = threadIdx.x & 31;
  int half = lane >> 4;
  int r    = lane & 15;
  int mrow = blockIdx.x * 16 + r;
  if (mrow >= n) mrow = n - 1;               // keep EXEC all-1s; stores are guarded
  int n0 = blockIdx.y * 32 + r;              // B tile rows (Wt is [Nn][K])
  const __bf16* arow  = Ab + (size_t)mrow * K;
  const __bf16* brow0 = Bb + (size_t)n0 * K;
  const __bf16* brow1 = brow0 + (size_t)16 * K;
  v8f c0 = {}, c1 = {};
  for (int k = 0; k < K; k += 32){
    v16bf a, b0, b1;
    int a1i = k + half * 16;                 // A: 16 contiguous K per lane half (matches B)
    int a2i = k + half * 16 + 8;
    int bbi = k + half * 16;                 // B: 16 contiguous K per lane half
#pragma unroll
    for (int i = 0; i < 8; i++){ a[i] = arow[a1i + i]; a[8 + i] = arow[a2i + i]; }
#pragma unroll
    for (int i = 0; i < 16; i++){ b0[i] = brow0[bbi + i]; }
#pragma unroll
    for (int i = 0; i < 16; i++){ b1[i] = brow1[bbi + i]; }
    c0 = __builtin_amdgcn_wmma_f32_16x16x32_bf16(false, a, false, b0, (short)0, c0, false, false);
    c1 = __builtin_amdgcn_wmma_f32_16x16x32_bf16(false, a, false, b1, (short)0, c1, false, false);
  }
  int col0 = blockIdx.y * 32 + r;
#pragma unroll
  for (int i = 0; i < 8; i++){
    int m = blockIdx.x * 16 + half * 8 + i;  // C/D 16x16 f32 layout
    if (m < n){
      C[(size_t)m * Nn + col0]      = c0[i];
      C[(size_t)m * Nn + col0 + 16] = c1[i];
    }
  }
}

// ---------------- attention scores: out[n][h] = h[n] . wv[:,h] ----------------
__global__ void att_scores_k(const float* __restrict__ h, const float* __restrict__ wv,
                             float* __restrict__ out, int n){
  int id = blockIdx.x * blockDim.x + threadIdx.x;
  if (id >= n * HEADSW) return;
  int node = id >> 2, hh = id & 3;
  const float* row = h + (size_t)node * HIDW;
  float s = 0.f;
  for (int k = 0; k < HIDW; k++) s += row[k] * wv[k * HEADSW + hh];
  out[id] = s;
}

__global__ void zero_k(unsigned* __restrict__ p, size_t n){
  size_t i = (size_t)blockIdx.x * blockDim.x + threadIdx.x;
  if (i < n) p[i] = 0u;
}

// ---------------- edge pass A: per-dst/head max of lrelu(a_s[src]+a_d[dst], 0.2) ----------------
__global__ void edge_max_k(const int* __restrict__ ei, int E, int nloop,
                           const float* __restrict__ a_s, const float* __restrict__ a_d,
                           unsigned* __restrict__ menc){
  int e = blockIdx.x * blockDim.x + threadIdx.x;
  if (e >= E + nloop) return;
  int src = (e < E) ? ei[e]     : (e - E);
  int dst = (e < E) ? ei[E + e] : (e - E);
#pragma unroll
  for (int hh = 0; hh < HEADSW; hh++){
    float al = lrelu_f(a_s[src * HEADSW + hh] + a_d[dst * HEADSW + hh], 0.2f);
    atomicMax(&menc[dst * HEADSW + hh], ord_enc(al));
  }
}

// ---------------- edge pass B: den[dst][h] += exp(alpha - m) ----------------
__global__ void edge_den_k(const int* __restrict__ ei, int E, int nloop,
                           const float* __restrict__ a_s, const float* __restrict__ a_d,
                           const unsigned* __restrict__ menc, float* __restrict__ den){
  int e = blockIdx.x * blockDim.x + threadIdx.x;
  if (e >= E + nloop) return;
  int src = (e < E) ? ei[e]     : (e - E);
  int dst = (e < E) ? ei[E + e] : (e - E);
#pragma unroll
  for (int hh = 0; hh < HEADSW; hh++){
    float al = lrelu_f(a_s[src * HEADSW + hh] + a_d[dst * HEADSW + hh], 0.2f);
    float m  = ord_dec(menc[dst * HEADSW + hh]);
    atomicAdd(&den[dst * HEADSW + hh], expf(al - m));
  }
}

// ---------------- edge pass C: wave per edge; nd[dst] += hs[src] * coef[head] ----------------
__global__ void edge_agg_k(const int* __restrict__ ei, int E, int nloop,
                           const float* __restrict__ a_s, const float* __restrict__ a_d,
                           const unsigned* __restrict__ menc, const float* __restrict__ den,
                           const float* __restrict__ hs, float* __restrict__ nd){
  int wid  = (int)(((size_t)blockIdx.x * blockDim.x + threadIdx.x) >> 5);
  int lane = threadIdx.x & 31;
  if (wid >= E + nloop) return;
  int src = (wid < E) ? ei[wid]     : (wid - E);
  int dst = (wid < E) ? ei[E + wid] : (wid - E);
  float coef[HEADSW];
#pragma unroll
  for (int hh = 0; hh < HEADSW; hh++){
    float al = lrelu_f(a_s[src * HEADSW + hh] + a_d[dst * HEADSW + hh], 0.2f);
    float m  = ord_dec(menc[dst * HEADSW + hh]);
    float ev = expf(al - m);
    coef[hh] = ev / (den[dst * HEADSW + hh] + 1e-16f);
  }
  const float* hrow = hs + (size_t)src * HIDW;
  float* orow = nd + (size_t)dst * HIDW;
#pragma unroll
  for (int t = 0; t < HEADSW; t++){
    int c = t * 32 + lane;                   // channel c -> head c/32 == t
    atomicAdd(&orow[c], hrow[c] * coef[t]);
  }
}

__global__ void add_bias_k(float* __restrict__ nd, const float* __restrict__ b, int n){
  int id = blockIdx.x * blockDim.x + threadIdx.x;
  if (id >= n * HIDW) return;
  nd[id] += b[id & (HIDW - 1)];
}

// ---------------- h = lrelu(new); refresh bf16 mirror ----------------
__global__ void fin_k(const float* __restrict__ nd, float* __restrict__ h,
                      unsigned short* __restrict__ hb, size_t n){
  size_t i = (size_t)blockIdx.x * blockDim.x + threadIdx.x;
  if (i >= n) return;
  float v = lrelu_f(nd[i], 0.01f);
  h[i] = v;
  hb[i] = f2bf(v);
}

// ---------------- heads: hid already = h@W1 (no bias); fuse b1+lrelu+W2 here ----------------
__global__ void lhead_k(const float* __restrict__ hid, const float* __restrict__ b1,
                        const float* __restrict__ W2, const float* __restrict__ b2,
                        float* __restrict__ mu, float* __restrict__ sg, int n){
  int node = blockIdx.x * blockDim.x + threadIdx.x;
  if (node >= n) return;
  float r0 = b2[0], r1 = b2[1], r2 = b2[2];
  const float* row = hid + (size_t)node * 32;
#pragma unroll
  for (int t = 0; t < 32; t++){
    float v = lrelu_f(row[t] + b1[t], 0.01f);
    r0 += v * W2[t * 3 + 0];
    r1 += v * W2[t * 3 + 1];
    r2 += v * W2[t * 3 + 2];
  }
  mu[(size_t)node * 2 + 0] = r0;
  mu[(size_t)node * 2 + 1] = r1;
  sg[node] = softplus_f(r2) + 1e-4f;
}

__global__ void rhead_k(const float* __restrict__ hid, const float* __restrict__ b1,
                        const float* __restrict__ W2, const float* __restrict__ b2,
                        float* __restrict__ mu, float* __restrict__ sg, int n){
  int node = blockIdx.x * blockDim.x + threadIdx.x;
  if (node >= n) return;
  float r0 = b2[0], r1 = b2[1];
  const float* row = hid + (size_t)node * 32;
#pragma unroll
  for (int t = 0; t < 32; t++){
    float v = lrelu_f(row[t] + b1[t], 0.01f);
    r0 += v * W2[t * 2 + 0];
    r1 += v * W2[t * 2 + 1];
  }
  mu[node] = r0;
  sg[node] = softplus_f(r1) + 1e-4f;
}

extern "C" void kernel_launch(void* const* d_in, const int* in_sizes, int n_in,
                              void* d_out, int out_size, void* d_ws, size_t ws_size,
                              hipStream_t stream){
  (void)n_in; (void)out_size; (void)ws_size;
  const int nl  = in_sizes[0] / 16;
  const int nr1 = in_sizes[1] / 8;
  const int nr2 = in_sizes[2] / 8;
  const int NTOT = nl + nr1 + nr2;
  const int ntoff[3] = {0, nl, nl + nr1};
  const int ntn[3]   = {nl, nr1, nr2};

  char* ws = (char*)d_ws;
  size_t off = 0;
  auto alloc = [&](size_t bytes)->void*{
    void* p = ws + off;
    off += (bytes + 255) & ~(size_t)255;
    return p;
  };
  float*          h    = (float*)          alloc((size_t)NTOT * HIDW * 4);
  unsigned short* hb   = (unsigned short*) alloc((size_t)NTOT * HIDW * 2);
  float*          nw   = (float*)          alloc((size_t)NTOT * HIDW * 4);
  float*          hs   = (float*)          alloc((size_t)nl * HIDW * 4);
  float*          as_  = (float*)          alloc((size_t)nl * HEADSW * 4);
  float*          ad_  = (float*)          alloc((size_t)nl * HEADSW * 4);
  unsigned*       menc = (unsigned*)       alloc((size_t)nl * HEADSW * 4);
  float*          den  = (float*)          alloc((size_t)nl * HEADSW * 4);
  unsigned short* wt   = (unsigned short*) alloc((size_t)HIDW * HIDW * 2);
  float*          was  = (float*)          alloc((size_t)HIDW * HEADSW * 4);
  float*          wad  = (float*)          alloc((size_t)HIDW * HEADSW * 4);
  unsigned short* w1t  = (unsigned short*) alloc((size_t)32 * HIDW * 2);
  float*          hid  = (float*)          alloc((size_t)nl * 32 * 4);

  // ---- encoders (param leaves: enc{lidar,radar1,radar2}{W,b,be,g} at 66/70/74) ----
  {
    const int encbase[3] = {66, 70, 74};
    const int indim[3]   = {16, 8, 8};
    for (int t = 0; t < 3; t++){
      enc_k<<<ntn[t], 128, 0, stream>>>((const float*)d_in[t], indim[t],
          (const float*)d_in[encbase[t] + 0], (const float*)d_in[encbase[t] + 1],
          (const float*)d_in[encbase[t] + 2], (const float*)d_in[encbase[t] + 3],
          h + (size_t)ntoff[t] * HIDW, hb + (size_t)ntoff[t] * HIDW, ntn[t]);
    }
  }

  // conv layer leaves sorted {ll,lr1,lr2,r1l,r1r1,r2l,r2r2}{W,ad,as,b}, layer0@10, layer1@38
  struct ET { int s, d, ei, p0, p1, self; };
  const ET ets[7] = {
    {0, 0, 3, 10, 38, 1},   // ll
    {1, 0, 4, 22, 50, 0},   // r1l
    {2, 0, 5, 30, 58, 0},   // r2l
    {0, 1, 6, 14, 42, 0},   // lr1
    {0, 2, 7, 18, 46, 0},   // lr2
    {1, 1, 8, 26, 54, 1},   // r1r1
    {2, 2, 9, 34, 62, 1},   // r2r2
  };

  for (int L = 0; L < 2; L++){
    {
      size_t n = (size_t)NTOT * HIDW;
      zero_k<<<(unsigned)((n + 255) / 256), 256, 0, stream>>>((unsigned*)nw, n);
    }
    for (int e = 0; e < 7; e++){
      const ET& et = ets[e];
      int pb = (L == 0) ? et.p0 : et.p1;
      const float* W  = (const float*)d_in[pb + 0];
      const float* ad = (const float*)d_in[pb + 1];
      const float* as = (const float*)d_in[pb + 2];
      const float* b  = (const float*)d_in[pb + 3];
      const int*   ei = (const int*)d_in[et.ei];
      int E  = in_sizes[et.ei] / 2;
      int ns = ntn[et.s], nd = ntn[et.d];
      int so = ntoff[et.s], dofs = ntoff[et.d];
      int nloop = et.self ? nd : 0;
      int tot = E + nloop;

      prep_wt_k<<<(HIDW * HIDW + 255) / 256, 256, 0, stream>>>(W, wt, HIDW, HIDW);
      prep_attvec_k<<<2, 256, 0, stream>>>(W, as, ad, was, wad);
      dim3 gg((ns + 15) / 16, HIDW / 32);
      gemm_wmma_k<<<gg, 32, 0, stream>>>(hb + (size_t)so * HIDW, wt, hs, ns, HIDW, HIDW);
      att_scores_k<<<(ns * HEADSW + 255) / 256, 256, 0, stream>>>(h + (size_t)so * HIDW, was, as_, ns);
      att_scores_k<<<(nd * HEADSW + 255) / 256, 256, 0, stream>>>(h + (size_t)dofs * HIDW, wad, ad_, nd);
      zero_k<<<(nd * HEADSW + 255) / 256, 256, 0, stream>>>(menc, (size_t)nd * HEADSW);
      zero_k<<<(nd * HEADSW + 255) / 256, 256, 0, stream>>>((unsigned*)den, (size_t)nd * HEADSW);
      edge_max_k<<<(tot + 255) / 256, 256, 0, stream>>>(ei, E, nloop, as_, ad_, menc);
      edge_den_k<<<(tot + 255) / 256, 256, 0, stream>>>(ei, E, nloop, as_, ad_, menc, den);
      edge_agg_k<<<(tot + 7) / 8, 256, 0, stream>>>(ei, E, nloop, as_, ad_, menc, den, hs,
                                                    nw + (size_t)dofs * HIDW);
      add_bias_k<<<(nd * HIDW + 255) / 256, 256, 0, stream>>>(nw + (size_t)dofs * HIDW, b, nd);
    }
    {
      size_t n = (size_t)NTOT * HIDW;
      fin_k<<<(unsigned)((n + 255) / 256), 256, 0, stream>>>(nw, h, hb, n);
    }
  }

  // ---- output heads ----
  float* out   = (float*)d_out;
  float* mu_l  = out;
  float* sg_l  = out + 2 * (size_t)nl;
  float* mu_r1 = out + 3 * (size_t)nl;
  float* s_r1  = mu_r1 + nr1;
  float* mu_r2 = s_r1 + nr1;
  float* s_r2  = mu_r2 + nr2;

  {   // l_head leaves: 78=W1(128x32) 79=W2(32x3) 80=b1 81=b2
    const float* W1 = (const float*)d_in[78];
    const float* W2 = (const float*)d_in[79];
    const float* b1 = (const float*)d_in[80];
    const float* b2 = (const float*)d_in[81];
    prep_wt_k<<<(HIDW * 32 + 255) / 256, 256, 0, stream>>>(W1, w1t, HIDW, 32);
    dim3 gg((nl + 15) / 16, 1);
    gemm_wmma_k<<<gg, 32, 0, stream>>>(hb, w1t, hid, nl, HIDW, 32);
    lhead_k<<<(nl + 255) / 256, 256, 0, stream>>>(hid, b1, W2, b2, mu_l, sg_l, nl);
  }
  {   // r_head leaves: 82=W1 83=W2(32x2) 84=b1 85=b2 (shared by both radar types)
    const float* W1 = (const float*)d_in[82];
    const float* W2 = (const float*)d_in[83];
    const float* b1 = (const float*)d_in[84];
    const float* b2 = (const float*)d_in[85];
    prep_wt_k<<<(HIDW * 32 + 255) / 256, 256, 0, stream>>>(W1, w1t, HIDW, 32);
    dim3 g1((nr1 + 15) / 16, 1);
    gemm_wmma_k<<<g1, 32, 0, stream>>>(hb + (size_t)nl * HIDW, w1t, hid, nr1, HIDW, 32);
    rhead_k<<<(nr1 + 255) / 256, 256, 0, stream>>>(hid, b1, W2, b2, mu_r1, s_r1, nr1);
    dim3 g2((nr2 + 15) / 16, 1);
    gemm_wmma_k<<<g2, 32, 0, stream>>>(hb + (size_t)(nl + nr1) * HIDW, w1t, hid, nr2, HIDW, 32);
    rhead_k<<<(nr2 + 255) / 256, 256, 0, stream>>>(hid, b1, W2, b2, mu_r2, s_r2, nr2);
  }
}
